// transformer_66632122630725
// MI455X (gfx1250) — matrix operand order repaired and checked by reading, not compiled, not run on hardware
//
#include <hip/hip_runtime.h>
#include <hip/hip_bf16.h>

// ---------------- model constants ----------------
#define BB      64
#define SEQL    1536
#define DM      256
#define NH      4
#define DHD     64
#define BKS     64
#define NBK     24
#define NLAYER  4
#define FFD     1024
#define MTOK    (BB * SEQL)   // 98304 tokens

typedef __attribute__((ext_vector_type(16))) _Float16 v16h;
typedef __attribute__((ext_vector_type(8)))  _Float16 v8h;
typedef __attribute__((ext_vector_type(8)))  float    v8f;
typedef unsigned int u32x4 __attribute__((ext_vector_type(4)));
typedef int          i32x8 __attribute__((ext_vector_type(8)));
typedef int          i32x4 __attribute__((ext_vector_type(4)));

// ---------------- WMMA helpers (CDNA5 16x16x32 f16 -> f32) ----------------
static __device__ __forceinline__ v8f wmma_f16(v16h a, v16h b, v8f c) {
  // (neg_a, A, neg_b, B, c_mod, C, reuse_a, reuse_b)
  return __builtin_amdgcn_wmma_f32_16x16x32_f16(false, a, false, b, (short)0, c,
                                                false, false);
}

// A fragment (16x32) from f16 row-major source: p = &A[row][k0 + (lane>>4)*16]
static __device__ __forceinline__ v16h ld_a_f16(const _Float16* p) {
  v8h lo = *reinterpret_cast<const v8h*>(p);
  v8h hi = *reinterpret_cast<const v8h*>(p + 8);
  v16h r;
#pragma unroll
  for (int i = 0; i < 8; ++i) { r[i] = lo[i]; r[8 + i] = hi[i]; }
  return r;
}

// B fragment (32x16) from Bt (=[N][K]) row-major f16: p = &Bt[n][k0 + (lane>>4)*16]
static __device__ __forceinline__ v16h ld_b_f16(const _Float16* p) {
  v8h lo = *reinterpret_cast<const v8h*>(p);
  v8h hi = *reinterpret_cast<const v8h*>(p + 8);
  v16h r;
#pragma unroll
  for (int i = 0; i < 8; ++i) { r[i] = lo[i]; r[8 + i] = hi[i]; }
  return r;
}

static __device__ __forceinline__ float gelu_tanh(float v) {
  return 0.5f * v * (1.0f + tanhf(0.7978845608028654f * (v + 0.044715f * v * v * v)));
}

// ---------------- Tensor Data Mover: contiguous f16 tile -> LDS ----------------
// 1-row tensor copy of `elems` 2-byte elements from gsrc to LDS offset lds_off.
// D# layout per CDNA5 ISA ch.8 (group0: count/lds/global/type, group1: dims).
// This toolchain exposes the 6-arg builtin: (g0, g1, g2, g3, g_extra, cpol).
static __device__ __forceinline__ void tdm_load_f16(const _Float16* gsrc,
                                                    unsigned lds_off,
                                                    unsigned elems) {
  unsigned long long ga = (unsigned long long)gsrc;
  u32x4 g0;
  g0[0] = 1u;                                   // count=1 (valid user descriptor)
  g0[1] = lds_off;                              // lds_addr (bytes)
  g0[2] = (unsigned)(ga & 0xFFFFFFFFu);         // global_addr[31:0]
  g0[3] = (unsigned)((ga >> 32) & 0x01FFFFFFu)  // global_addr[56:32]
          | 0x80000000u;                        // type=2 ("image") in bits 127:126
  i32x8 g1;
  g1[0] = 0x00010000;                           // wg_mask=0, data_size=1 (2 bytes)
  g1[1] = (int)((elems & 0xFFFFu) << 16);       // tensor_dim0[15:0] at bits 63:48
  g1[2] = (int)(((elems >> 16) & 0xFFFFu)       // tensor_dim0[31:16]
                | (1u << 16));                  // tensor_dim1 = 1
  g1[3] = (int)((elems & 0xFFFFu) << 16);       // tile_dim0 at bits 127:112
  g1[4] = 1;                                    // tile_dim1 = 1, tile_dim2 = 0
  g1[5] = (int)elems;                           // tensor_dim0_stride[31:0]
  g1[6] = 0;
  g1[7] = 0;
  i32x4 z4 = {0, 0, 0, 0};
  i32x8 z8 = {0, 0, 0, 0, 0, 0, 0, 0};
  __builtin_amdgcn_tensor_load_to_lds(g0, g1, z4, z4, z8, 0);
}

// ---------------- weight f32[K][N] -> f16 transposed [N][K] ----------------
__global__ void conv_t_kernel(const float* __restrict__ in, _Float16* __restrict__ out,
                              int K, int N) {
  int i = blockIdx.x * 256 + threadIdx.x;
  if (i < K * N) {
    int k = i / N, n = i % N;
    out[n * K + k] = (_Float16)in[i];
  }
}

// ---------------- embedding (token + axial pos) ----------------
__global__ void embed_kernel(const int* __restrict__ tokens, const float* __restrict__ emb,
                             const float* __restrict__ prow, const float* __restrict__ pcol,
                             float* __restrict__ x1, float* __restrict__ x2) {
  long t = blockIdx.x;
  int d = threadIdx.x;
  int s = (int)(t % SEQL);
  int tok = tokens[t];
  float v = emb[tok * DM + d] + prow[(s >> 6) * DM + d] + pcol[(s & 63) * DM + d];
  x1[t * DM + d] = v;
  x2[t * DM + d] = v;
}

// ---------------- LayerNorm: one wave per token, f16 output ----------------
__global__ __launch_bounds__(256) void ln_kernel(const float* __restrict__ x,
                                                 _Float16* __restrict__ y,
                                                 const float* __restrict__ g,
                                                 const float* __restrict__ bta) {
  int wave = threadIdx.x >> 5, lane = threadIdx.x & 31;
  long t = (long)blockIdx.x * 8 + wave;
  const float* xr = x + t * DM;
  float v[8];
  float s = 0.f;
#pragma unroll
  for (int i = 0; i < 8; ++i) { v[i] = xr[lane + i * 32]; s += v[i]; }
#pragma unroll
  for (int d = 1; d < 32; d <<= 1) s += __shfl_xor(s, d, 32);
  float mean = s * (1.0f / DM);
  float q = 0.f;
#pragma unroll
  for (int i = 0; i < 8; ++i) { float dd = v[i] - mean; q += dd * dd; }
#pragma unroll
  for (int d = 1; d < 32; d <<= 1) q += __shfl_xor(q, d, 32);
  float rstd = rsqrtf(q * (1.0f / DM) + 1e-5f);
#pragma unroll
  for (int i = 0; i < 8; ++i) {
    int c = lane + i * 32;
    y[t * DM + c] = (_Float16)((v[i] - mean) * rstd * g[c] + bta[c]);
  }
}

// ---------------- generic GEMM: C[M,N] = A[M,K](f16) * Bt[N,K](f16) ----------------
// EPI 0: store f16.  EPI 1: out(f32) += acc + bias[n]  (residual add)
template <int EPI>
__global__ __launch_bounds__(256) void gemm_kernel(const _Float16* __restrict__ A,
                                                   const _Float16* __restrict__ BT,
                                                   void* __restrict__ outp,
                                                   const float* __restrict__ bias,
                                                   int N, int K) {
  const int w = threadIdx.x >> 5;
  const int lane = threadIdx.x & 31;
  const int lanelow = lane & 15, hic = lane >> 4;
  const int wm = w & 3, wn = w >> 2;
  const long rowBase = (long)blockIdx.x * 256 + wm * 64;
  const int colBase = blockIdx.y * 128 + wn * 64;

  v8f acc[4][4] = {};
  for (int k0 = 0; k0 < K; k0 += 32) {
    v16h a[4], b[4];
#pragma unroll
    for (int ms = 0; ms < 4; ++ms)
      a[ms] = ld_a_f16(A + (rowBase + ms * 16 + lanelow) * (long)K + k0 + hic * 16);
#pragma unroll
    for (int ns = 0; ns < 4; ++ns)
      b[ns] = ld_b_f16(BT + (long)(colBase + ns * 16 + lanelow) * K + k0 + hic * 16);
#pragma unroll
    for (int ms = 0; ms < 4; ++ms)
#pragma unroll
      for (int ns = 0; ns < 4; ++ns)
        acc[ms][ns] = wmma_f16(a[ms], b[ns], acc[ms][ns]);
  }
#pragma unroll
  for (int ms = 0; ms < 4; ++ms)
#pragma unroll
    for (int ns = 0; ns < 4; ++ns)
#pragma unroll
      for (int r = 0; r < 8; ++r) {
        long row = rowBase + ms * 16 + r + hic * 8;
        int col = colBase + ns * 16 + lanelow;
        if (EPI == 0) {
          ((_Float16*)outp)[row * N + col] = (_Float16)acc[ms][ns][r];
        } else {
          float* o = (float*)outp;
          o[row * N + col] += acc[ms][ns][r] + bias[col];
        }
      }
}

// ---------------- AttentionSortNet: bucket means, R, softmax/argmax ----------------
__global__ __launch_bounds__(256) void sortnet_kernel(const _Float16* __restrict__ q16,
                                                      const _Float16* __restrict__ kv16,
                                                      int* __restrict__ idxb,
                                                      float* __restrict__ svalb) {
  __shared__ float sq[NBK * DHD];
  __shared__ float sk[NBK * DHD];
  __shared__ float Rm[NBK * NBK];
  int bh = blockIdx.x;
  int b = bh >> 2, h = bh & 3;
  long base = (long)b * SEQL;
  for (int p = threadIdx.x; p < NBK * DHD; p += 256) {
    int u = p >> 6, d = p & 63;
    float aq = 0.f, ak = 0.f;
    for (int t = 0; t < BKS; ++t) {
      long row = base + u * BKS + t;
      aq += (float)q16[row * 256 + h * DHD + d];
      ak += (float)kv16[row * 512 + h * DHD + d];
    }
    sq[p] = aq * (1.0f / BKS);
    sk[p] = ak * (1.0f / BKS);
  }
  __syncthreads();
  for (int p = threadIdx.x; p < NBK * NBK; p += 256) {
    int u = p / NBK, vv = p % NBK;
    float dsum = 0.f;
    for (int d = 0; d < DHD; ++d) dsum += sq[u * DHD + d] * sk[vv * DHD + d];
    Rm[p] = dsum * 0.75f;  // * TEMP
  }
  __syncthreads();
  if (threadIdx.x < NBK) {
    int u = threadIdx.x;
    float mx = -3.0e38f; int arg = 0;
    for (int vv = 0; vv < NBK; ++vv) {
      float r = Rm[u * NBK + vv];
      if (r > mx) { mx = r; arg = vv; }
    }
    float ssum = 0.f;
    for (int vv = 0; vv < NBK; ++vv) ssum += __expf(Rm[u * NBK + vv] - mx);
    idxb[bh * NBK + u] = arg;
    svalb[bh * NBK + u] = 1.0f / ssum;  // soft value at the argmax
  }
}

// ---------------- bucket attention: one block per (b,h,bucket), f16 out ----------------
__global__ __launch_bounds__(128) void attn_kernel(const _Float16* __restrict__ q16,
                                                   const _Float16* __restrict__ kv16,
                                                   const int* __restrict__ tokens,
                                                   const int* __restrict__ idxb,
                                                   const float* __restrict__ svalb,
                                                   _Float16* __restrict__ o16) {
  constexpr int PSS = 136;  // 128 + 8 pad halfs (keeps 16B alignment)
  __shared__ __align__(16) _Float16 vts[DHD * PSS];  // v^T  [d][j]
  __shared__ __align__(16) _Float16 ps[64 * PSS];    // probs [i][j]
  __shared__ int mq[64];
  __shared__ int mk[128];

  int blk = blockIdx.x;
  int u = blk % NBK;
  int bh = blk / NBK;
  int h = bh & 3, b = bh >> 2;
  int idxu = idxb[bh * NBK + u];
  float s = svalb[bh * NBK + u];
  long tokQ = (long)b * SEQL + u * BKS;
  long tokR = (long)b * SEQL + idxu * BKS;

  // masks + transposed values into LDS
  for (int p = threadIdx.x; p < 64; p += 128) mq[p] = tokens[tokQ + p] > 0;
  for (int p = threadIdx.x; p < 128; p += 128) {
    long tr = (p < 64) ? (tokR + p) : (tokQ + (p - 64));
    mk[p] = tokens[tr] > 0;
  }
  for (int p = threadIdx.x; p < 64 * 128; p += 128) {
    int j = p >> 6, d = p & 63;
    long tr = (j < 64) ? (tokR + j) : (tokQ + (j - 64));
    vts[d * PSS + j] = kv16[tr * 512 + 256 + h * DHD + d];
  }

  const int w = threadIdx.x >> 5;
  const int lane = threadIdx.x & 31;
  const int lanelow = lane & 15, hic = lane >> 4;

  // dots: rows [w*16, w*16+16) x 128 keys, K = DH = 64 (k fragments straight from global)
  v8f acc[8] = {};
  for (int k0 = 0; k0 < DHD; k0 += 32) {
    v16h a = ld_a_f16(q16 + (tokQ + w * 16 + lanelow) * 256 + h * DHD + k0 + hic * 16);
#pragma unroll
    for (int f = 0; f < 8; ++f) {
      int j = f * 16 + lanelow;
      long tr = (j < 64) ? (tokR + j) : (tokQ + (j - 64));
      v16h bb = ld_b_f16(kv16 + tr * 512 + h * DHD + k0 + hic * 16);
      acc[f] = wmma_f16(a, bb, acc[f]);
    }
  }
  __syncthreads();  // masks + vts visible

  // in-register softmax per row (rows of one wave live in 16-lane halves)
  const float invDH = 0.125f;  // DH^-0.5
#pragma unroll
  for (int r = 0; r < 8; ++r) {
    int mrow = w * 16 + r + hic * 8;
    bool qm = mq[mrow] != 0;
    float vals[8];
    float vmax = -3.0e38f;
#pragma unroll
    for (int f = 0; f < 8; ++f) {
      int j = f * 16 + lanelow;
      float v = acc[f][r] * invDH;
      if (f < 4) v *= s;              // routing scale on routed half (j<64)
      if (!qm || !mk[j]) v = -1.0e9f;
      vals[f] = v;
      vmax = fmaxf(vmax, v);
    }
#pragma unroll
    for (int t = 1; t < 16; t <<= 1) vmax = fmaxf(vmax, __shfl_xor(vmax, t, 32));
    float ssum = 0.f;
#pragma unroll
    for (int f = 0; f < 8; ++f) { vals[f] = __expf(vals[f] - vmax); ssum += vals[f]; }
#pragma unroll
    for (int t = 1; t < 16; t <<= 1) ssum += __shfl_xor(ssum, t, 32);
    float inv = 1.0f / ssum;
#pragma unroll
    for (int f = 0; f < 8; ++f) {
      int j = f * 16 + lanelow;
      float p = vals[f] * inv;
      if (f < 4) p *= s;              // routing scale applied to routed V half
      ps[mrow * PSS + j] = (_Float16)p;
    }
  }
  __syncthreads();  // ps ready

  // o = probs @ v2 : A = ps (K=128), B = vts (v^T)
  v8f oacc[4] = {};
  for (int k0 = 0; k0 < 128; k0 += 32) {
    v16h a = ld_a_f16(ps + (w * 16 + lanelow) * PSS + k0 + hic * 16);
#pragma unroll
    for (int ns = 0; ns < 4; ++ns) {
      v16h bb = ld_b_f16(vts + (ns * 16 + lanelow) * PSS + k0 + hic * 16);
      oacc[ns] = wmma_f16(a, bb, oacc[ns]);
    }
  }
#pragma unroll
  for (int ns = 0; ns < 4; ++ns)
#pragma unroll
    for (int r = 0; r < 8; ++r) {
      long row = tokQ + w * 16 + r + hic * 8;
      o16[row * 256 + h * DHD + ns * 16 + lanelow] = (_Float16)oacc[ns][r];
    }
}

// ---------------- fused FFN: x2 += gelu(xln@w1+b1)@w2 + b2 ----------------
// w1T chunks staged into LDS by the Tensor Data Mover (TENSORcnt-tracked).
__global__ __launch_bounds__(256) void ffn_kernel(const _Float16* __restrict__ xln,
                                                  float* __restrict__ x2,
                                                  const _Float16* __restrict__ w1T,
                                                  const _Float16* __restrict__ w2T,
                                                  const float* __restrict__ b1,
                                                  const float* __restrict__ b2) {
  constexpr int HSS = 72;  // 64 + 8 pad halfs
  __shared__ __align__(16) _Float16 hs[128 * HSS];
  __shared__ __align__(16) _Float16 w1buf[64 * DM];  // 32 KB TDM-staged w1T chunk
  const int w = threadIdx.x >> 5;
  const int lane = threadIdx.x & 31;
  const int lanelow = lane & 15, hic = lane >> 4;
  const int wm = w & 1, wn = w >> 1;
  const long mBase = (long)blockIdx.x * 128;
  const unsigned w1buf_lds = (unsigned)(size_t)(&w1buf[0]);  // LDS offset = low 32 bits

  v8f acc[4][4] = {};  // output tile 128x256, this wave: rows wm*64.., cols wn*64..

  for (int c = 0; c < FFD; c += 64) {
    __syncthreads();  // all readers of w1buf/hs from previous chunk are done
    if (w == 0) {     // one TDM issue per workgroup (EXEC ignored by TDM)
      tdm_load_f16(w1T + (size_t)c * DM, w1buf_lds, 64 * DM);
      __builtin_amdgcn_s_wait_tensorcnt(0);
    }
    __syncthreads();  // w1buf ready

    // ---- h = gelu(xln @ w1T[c:c+64] + b1) ; this wave: rows w*16..w*16+15
    v8f hacc[4] = {};
    for (int k0 = 0; k0 < DM; k0 += 32) {
      v16h a = ld_a_f16(xln + (mBase + w * 16 + lanelow) * DM + k0 + hic * 16);
#pragma unroll
      for (int ns = 0; ns < 4; ++ns) {
        v16h bb = ld_b_f16(w1buf + (ns * 16 + lanelow) * DM + k0 + hic * 16);
        hacc[ns] = wmma_f16(a, bb, hacc[ns]);
      }
    }
#pragma unroll
    for (int ns = 0; ns < 4; ++ns)
#pragma unroll
      for (int r = 0; r < 8; ++r) {
        int ml = w * 16 + r + hic * 8;
        int nl = ns * 16 + lanelow;
        float v = hacc[ns][r] + b1[c + nl];
        hs[ml * HSS + nl] = (_Float16)gelu_tanh(v);
      }
    __syncthreads();  // hs ready

    // ---- acc += h @ w2T[:, c:c+64]
    for (int k0 = 0; k0 < 64; k0 += 32) {
      v16h a2[4];
#pragma unroll
      for (int ms = 0; ms < 4; ++ms)
        a2[ms] = ld_a_f16(hs + (wm * 64 + ms * 16 + lanelow) * HSS + k0 + hic * 16);
#pragma unroll
      for (int ns = 0; ns < 4; ++ns) {
        v16h bb = ld_b_f16(w2T + (long)(wn * 64 + ns * 16 + lanelow) * FFD + c + k0 + hic * 16);
#pragma unroll
        for (int ms = 0; ms < 4; ++ms)
          acc[ms][ns] = wmma_f16(a2[ms], bb, acc[ms][ns]);
      }
    }
  }

#pragma unroll
  for (int ms = 0; ms < 4; ++ms)
#pragma unroll
    for (int ns = 0; ns < 4; ++ns)
#pragma unroll
      for (int r = 0; r < 8; ++r) {
        long row = mBase + wm * 64 + ms * 16 + r + hic * 8;
        int col = wn * 64 + ns * 16 + lanelow;
        x2[row * DM + col] += acc[ms][ns][r] + b2[col];
      }
}

// ---------------- masked mean pool over sequence ----------------
__global__ void pool_kernel(const float* __restrict__ x1, const float* __restrict__ x2,
                            const int* __restrict__ tokens, float* __restrict__ out) {
  int b = blockIdx.x, d = threadIdx.x;
  float acc = 0.f;
  int cnt = 0;
  for (int sidx = 0; sidx < SEQL; ++sidx) {
    long t = (long)b * SEQL + sidx;
    if (tokens[t] > 0) {
      acc += 0.5f * (x1[t * DM + d] + x2[t * DM + d]);
      ++cnt;
    }
  }
  out[b * DM + d] = cnt ? acc / (float)cnt : 0.0f;
}

// ---------------- host orchestration ----------------
extern "C" void kernel_launch(void* const* d_in, const int* in_sizes, int n_in,
                              void* d_out, int out_size, void* d_ws, size_t ws_size,
                              hipStream_t stream) {
  (void)in_sizes; (void)n_in; (void)out_size; (void)ws_size;
  const int*   tokens  = (const int*)d_in[0];
  const float* tok_emb = (const float*)d_in[1];
  const float* pos_row = (const float*)d_in[2];
  const float* pos_col = (const float*)d_in[3];
  const float* wq   = (const float*)d_in[4];
  const float* wkv  = (const float*)d_in[5];
  const float* wo   = (const float*)d_in[6];
  const float* bo   = (const float*)d_in[7];
  const float* ln1g = (const float*)d_in[8];
  const float* ln1b = (const float*)d_in[9];
  const float* w1   = (const float*)d_in[10];
  const float* b1   = (const float*)d_in[11];
  const float* w2   = (const float*)d_in[12];
  const float* b2   = (const float*)d_in[13];
  const float* ln2g = (const float*)d_in[14];
  const float* ln2b = (const float*)d_in[15];

  char* ws = (char*)d_ws;
  size_t off = 0;
  auto alloc = [&](size_t bytes) -> void* {
    off = (off + 255) & ~(size_t)255;
    void* p = ws + off;
    off += bytes;
    return p;
  };
  float*    x1    = (float*)alloc((size_t)MTOK * DM * 4);
  float*    x2    = (float*)alloc((size_t)MTOK * DM * 4);
  _Float16* xln16 = (_Float16*)alloc((size_t)MTOK * DM * 2);
  _Float16* o16   = (_Float16*)alloc((size_t)MTOK * DM * 2);
  _Float16* q16   = (_Float16*)alloc((size_t)MTOK * DM * 2);
  _Float16* kv16  = (_Float16*)alloc((size_t)MTOK * 2 * DM * 2);
  _Float16* w16   = (_Float16*)alloc((size_t)NLAYER * 786432 * 2);
  int*      idxb  = (int*)alloc(256 * NBK * 4);
  float*    svalb = (float*)alloc(256 * NBK * 4);

  // per-layer f16 transposed weight offsets (halfs)
  // wqT:0 (64K)  wkvT:65536 (128K)  woT:196608 (64K)  w1T:262144 (256K)  w2T:524288 (256K)
  for (int l = 0; l < NLAYER; ++l) {
    _Float16* wl = w16 + (size_t)l * 786432;
    conv_t_kernel<<<(256 * 256 + 255) / 256, 256, 0, stream>>>(
        wq + (size_t)l * 256 * 256, wl + 0, 256, 256);
    conv_t_kernel<<<(256 * 512 + 255) / 256, 256, 0, stream>>>(
        wkv + (size_t)l * 256 * 512, wl + 65536, 256, 512);
    conv_t_kernel<<<(256 * 256 + 255) / 256, 256, 0, stream>>>(
        wo + (size_t)l * 256 * 256, wl + 196608, 256, 256);
    conv_t_kernel<<<(256 * 1024 + 255) / 256, 256, 0, stream>>>(
        w1 + (size_t)l * 256 * 1024, wl + 262144, 256, 1024);
    conv_t_kernel<<<(1024 * 256 + 255) / 256, 256, 0, stream>>>(
        w2 + (size_t)l * 1024 * 256, wl + 524288, 1024, 256);
  }

  embed_kernel<<<MTOK, 256, 0, stream>>>(tokens, tok_emb, pos_row, pos_col, x1, x2);

  for (int l = 0; l < NLAYER; ++l) {
    const _Float16* wl = w16 + (size_t)l * 786432;
    // y1 = x1 + attn(LN1(x2))
    ln_kernel<<<MTOK / 8, 256, 0, stream>>>(x2, xln16, ln1g + l * DM, ln1b + l * DM);
    gemm_kernel<0><<<dim3(MTOK / 256, 2), 256, 0, stream>>>(xln16, wl + 0, q16, nullptr, 256, 256);
    gemm_kernel<0><<<dim3(MTOK / 256, 4), 256, 0, stream>>>(xln16, wl + 65536, kv16, nullptr, 512, 256);
    sortnet_kernel<<<BB * NH, 256, 0, stream>>>(q16, kv16, idxb, svalb);
    attn_kernel<<<BB * NH * NBK, 128, 0, stream>>>(q16, kv16, tokens, idxb, svalb, o16);
    gemm_kernel<1><<<dim3(MTOK / 256, 2), 256, 0, stream>>>(o16, wl + 196608, x1, bo + l * DM, 256, 256);
    // x2 = x2 + ff(LN2(x1))
    ln_kernel<<<MTOK / 8, 256, 0, stream>>>(x1, xln16, ln2g + l * DM, ln2b + l * DM);
    ffn_kernel<<<MTOK / 128, 256, 0, stream>>>(xln16, x2, wl + 262144, wl + 524288,
                                               b1 + (size_t)l * FFD, b2 + l * DM);
  }

  pool_kernel<<<BB, 256, 0, stream>>>(x1, x2, tokens, (float*)d_out);
}
